// CellHeaderAttentionEncoder_53910429499715
// MI455X (gfx1250) — compile-verified, matrix operand order checked by reading, not run on hardware
//
#include <hip/hip_runtime.h>
#include <hip/hip_bf16.h>

#define HIDDEN   768
#define NHEADS   8
#define DH       96
#define MAXH     8

typedef __attribute__((ext_vector_type(16))) _Float16 v16h;
typedef __attribute__((ext_vector_type(8)))  _Float16 v8h;
typedef __attribute__((ext_vector_type(8)))  float    v8f;

// ---------------------------------------------------------------------------
// Load a 16x32 f16 fragment (A or B operand of v_wmma_f32_16x16x32_f16) from a
// row-major [rows x ld] f16 matrix. Per CDNA5 ISA 7.12.2 (16-bit A 16x32):
//   lane l (0..15): row M=l, K=0..7 in v[0..3], K=16..23 in v[4..7]
//   lane l+16:      row M=l, K=8..15 / K=24..31
// => per lane: base + m*ld + 8*(lane>>4), take 8 contiguous + 8 at +16.
// ---------------------------------------------------------------------------
__device__ __forceinline__ v16h load_frag16(const _Float16* __restrict__ base, int ld) {
    const int lane = threadIdx.x & 31;
    const int m    = lane & 15;
    const int koff = (lane >> 4) << 3;           // 0 or 8
    const _Float16* p = base + (size_t)m * ld + koff;
    v8h lo = *(const v8h*)(p);
    v8h hi = *(const v8h*)(p + 16);
    v16h f;
#pragma unroll
    for (int j = 0; j < 8; ++j) { f[j] = lo[j]; f[j + 8] = hi[j]; }
    return f;
}

// ---------------------------------------------------------------------------
// f32 -> f16 convert
// ---------------------------------------------------------------------------
__global__ void cvt_f16_kernel(const float* __restrict__ src, _Float16* __restrict__ dst, int n) {
    int i = blockIdx.x * blockDim.x + threadIdx.x;
    if (i < n) dst[i] = (_Float16)src[i];
}

// ---------------------------------------------------------------------------
// cwp = cell + row_emb[clip(r)] + col_emb[clip(c)]   (f32 + f16 copies)
// one block per cell row
// ---------------------------------------------------------------------------
__global__ void prep_kernel(const float* __restrict__ cell,
                            const float* __restrict__ row_emb,
                            const float* __restrict__ col_emb,
                            const int*   __restrict__ pos,
                            float* __restrict__ cwp_f,
                            _Float16* __restrict__ cwp_h) {
    const int n = blockIdx.x;
    int r = pos[n * 2 + 0]; r = r < 0 ? 0 : (r > 99 ? 99 : r);
    int c = pos[n * 2 + 1]; c = c < 0 ? 0 : (c > 99 ? 99 : c);
    const size_t nb = (size_t)n * HIDDEN;
    for (int d = threadIdx.x; d < HIDDEN; d += blockDim.x) {
        float v = cell[nb + d] + row_emb[(size_t)r * HIDDEN + d] + col_emb[(size_t)c * HIDDEN + d];
        cwp_f[nb + d] = v;
        cwp_h[nb + d] = (_Float16)v;
    }
}

// ---------------------------------------------------------------------------
// Generic WMMA GEMM:  C[M,Nout] = A[M,K] @ B[Nout,K]^T + bias[Nout]
// Block = 256 threads = 8 waves, block tile 64 rows x 128 cols,
// each wave owns a 16x64 tile (4 accumulators), K-loop step 32.
// All 10 global_load_b128 of a step are issued before the step's 4 WMMAs so
// the waits are staggered; unroll 2 lets the scheduler overlap iterations.
// ---------------------------------------------------------------------------
template <bool OUT_F16>
__global__ void gemm_kernel(const _Float16* __restrict__ A,
                            const _Float16* __restrict__ B,
                            const float*    __restrict__ bias,
                            void*           __restrict__ C,
                            int Nout, int K, int lda, int ldb) {
    const int wave = threadIdx.x >> 5;
    const int lane = threadIdx.x & 31;
    const int row0 = blockIdx.x * 64 + (wave & 3) * 16;
    const int col0 = blockIdx.y * 128 + (wave >> 2) * 64;

    v8f acc[4];
#pragma unroll
    for (int t = 0; t < 4; ++t) {
        float b = bias[col0 + t * 16 + (lane & 15)];
#pragma unroll
        for (int j = 0; j < 8; ++j) acc[t][j] = b;
    }

    const _Float16* Arow = A + (size_t)row0 * lda;
    const _Float16* Bcol = B + (size_t)col0 * ldb;

#pragma unroll 2
    for (int k = 0; k < K; k += 32) {
        v16h a = load_frag16(Arow + k, lda);
        v16h b[4];
#pragma unroll
        for (int t = 0; t < 4; ++t)
            b[t] = load_frag16(Bcol + (size_t)t * 16 * ldb + k, ldb);
#pragma unroll
        for (int t = 0; t < 4; ++t)
            acc[t] = __builtin_amdgcn_wmma_f32_16x16x32_f16(
                false, a, false, b[t], (short)0, acc[t], false, false);
    }

    // D layout: vgpr r, lanes 0-15 -> row r, lanes 16-31 -> row r+8; col = lane&15
    const int rb = row0 + ((lane >> 4) << 3);
#pragma unroll
    for (int t = 0; t < 4; ++t) {
        const int ncol = col0 + t * 16 + (lane & 15);
#pragma unroll
        for (int r = 0; r < 8; ++r) {
            size_t idx = (size_t)(rb + r) * Nout + ncol;
            if (OUT_F16) ((_Float16*)C)[idx] = (_Float16)acc[t][r];
            else         ((float*)C)[idx]    = acc[t][r];
        }
    }
}

// ---------------------------------------------------------------------------
// Per-cell attention over up to 8 gathered (projected) headers.
// One 256-thread block per cell. khdr/vhdr are tiny (1.5 MB) -> cache-resident.
// ---------------------------------------------------------------------------
__global__ void attn_kernel(const _Float16* __restrict__ q_h,
                            const float* __restrict__ khdr,
                            const float* __restrict__ vhdr,
                            const int*   __restrict__ ids_map,
                            _Float16* __restrict__ ctx_h,
                            float* __restrict__ weights_out) {
    const int n = blockIdx.x;
    const int t = threadIdx.x;
    __shared__ float sc[8][8];     // scores [head a][header h]
    __shared__ float at[8][8];     // attn
    __shared__ int   sid[8];
    __shared__ float smask[8];

    if (t < 8) {
        int id = ids_map[n * MAXH + t];
        sid[t]   = id >= 0 ? id : 0;
        smask[t] = id >= 0 ? 1.f : 0.f;
    }
    __syncthreads();

    if (t < 64) {
        const int a = t >> 3, h = t & 7;
        const _Float16* qp = q_h + (size_t)n * HIDDEN + a * DH;
        const float*    kp = khdr + (size_t)sid[h] * HIDDEN + a * DH;
        float d = 0.f;
#pragma unroll 8
        for (int j = 0; j < DH; ++j) d += (float)qp[j] * kp[j];
        sc[a][h] = d * 0.10206207261596575f;   // 1/sqrt(96)
    }
    __syncthreads();

    if (t < 8) {
        const int a = t;
        float s[8], mx = -1e30f;
#pragma unroll
        for (int h = 0; h < 8; ++h) {
            float v = smask[h] > 0.f ? sc[a][h] : -1e9f;
            s[h] = v; mx = fmaxf(mx, v);
        }
        float sum = 0.f;
#pragma unroll
        for (int h = 0; h < 8; ++h) { s[h] = __expf(s[h] - mx); sum += s[h]; }
        float inv = 1.f / sum;
#pragma unroll
        for (int h = 0; h < 8; ++h) at[a][h] = smask[h] > 0.f ? s[h] * inv : 0.f;
    }
    __syncthreads();

    for (int d = t; d < HIDDEN; d += 256) {
        const int a = d / DH;
        float c = 0.f;
#pragma unroll
        for (int h = 0; h < 8; ++h) c += at[a][h] * vhdr[(size_t)sid[h] * HIDDEN + d];
        ctx_h[(size_t)n * HIDDEN + d] = (_Float16)c;
    }
    if (t < 8) {
        float m = 0.f;
#pragma unroll
        for (int a = 0; a < 8; ++a) m += at[a][t];
        weights_out[(size_t)n * MAXH + t] = smask[t] > 0.f ? m * 0.125f : 0.f;
    }
}

// ---------------------------------------------------------------------------
// Fusion: h = cwp @ fus_w[:, :768]^T + att_out @ fus_w[:, 768:]^T + fus_b
// followed in-kernel by LayerNorm + ReLU + mask-blend into d_out.
// Block = 512 threads (16 waves) owns 16 full rows (16 x 768 output).
// Wave w computes cols [48w, 48w+48) -> 3 WMMA tiles (low register pressure,
// no spills); LN epilogue: wave w normalizes row w.
// ---------------------------------------------------------------------------
__global__ void fusion_kernel(const _Float16* __restrict__ cwp_h,
                              const _Float16* __restrict__ attout_h,
                              const _Float16* __restrict__ w_fus_h,   // [768][1536]
                              const float* __restrict__ fus_b,
                              const float* __restrict__ ln_g,
                              const float* __restrict__ ln_b,
                              const float* __restrict__ cwp_f,
                              const int*   __restrict__ ids_map,
                              float* __restrict__ out) {
    __shared__ float hbuf[16][HIDDEN + 8];

    const int wave = threadIdx.x >> 5;      // 0..15
    const int lane = threadIdx.x & 31;
    const int row0 = blockIdx.x * 16;
    const int col0 = wave * 48;

    v8f acc[3];
#pragma unroll
    for (int t = 0; t < 3; ++t) {
        float b = fus_b[col0 + t * 16 + (lane & 15)];
#pragma unroll
        for (int j = 0; j < 8; ++j) acc[t][j] = b;
    }

#pragma unroll
    for (int phase = 0; phase < 2; ++phase) {
        const _Float16* A    = phase ? attout_h : cwp_h;
        const _Float16* Bb   = w_fus_h + phase * HIDDEN;   // column half of fus_w
        const _Float16* Arow = A + (size_t)row0 * HIDDEN;
        const _Float16* Bcol = Bb + (size_t)col0 * (2 * HIDDEN);

#pragma unroll 2
        for (int k = 0; k < HIDDEN; k += 32) {
            v16h a = load_frag16(Arow + k, HIDDEN);
            v16h b[3];
#pragma unroll
            for (int t = 0; t < 3; ++t)
                b[t] = load_frag16(Bcol + (size_t)t * 16 * (2 * HIDDEN) + k, 2 * HIDDEN);
#pragma unroll
            for (int t = 0; t < 3; ++t)
                acc[t] = __builtin_amdgcn_wmma_f32_16x16x32_f16(
                    false, a, false, b[t], (short)0, acc[t], false, false);
        }
    }

    // spill 16x48 wave tile into LDS row buffer
    const int lrow = (lane >> 4) << 3;
    const int lcol = lane & 15;
#pragma unroll
    for (int t = 0; t < 3; ++t)
#pragma unroll
        for (int r = 0; r < 8; ++r)
            hbuf[lrow + r][col0 + t * 16 + lcol] = acc[t][r];
    __syncthreads();

    // LayerNorm + ReLU + blend; wave w handles row w
    {
        const int r = wave;
        const int n = row0 + r;
        float s = 0.f, s2 = 0.f;
        for (int d = lane; d < HIDDEN; d += 32) {
            float x = hbuf[r][d];
            s += x; s2 += x * x;
        }
#pragma unroll
        for (int o = 16; o > 0; o >>= 1) {
            s  += __shfl_xor(s,  o, 32);
            s2 += __shfl_xor(s2, o, 32);
        }
        const float mu   = s * (1.f / HIDDEN);
        const float var  = s2 * (1.f / HIDDEN) - mu * mu;
        const float rinv = rsqrtf(var + 1e-5f);

        bool has = false;
#pragma unroll
        for (int h = 0; h < 8; ++h) has |= (ids_map[(size_t)n * MAXH + h] >= 0);

        for (int d = lane; d < HIDDEN; d += 32) {
            float x = (hbuf[r][d] - mu) * rinv * ln_g[d] + ln_b[d];
            x = x > 0.f ? x : 0.f;
            out[(size_t)n * HIDDEN + d] = has ? x : cwp_f[(size_t)n * HIDDEN + d];
        }
    }
}

// ---------------------------------------------------------------------------
extern "C" void kernel_launch(void* const* d_in, const int* in_sizes, int n_in,
                              void* d_out, int out_size, void* d_ws, size_t ws_size,
                              hipStream_t stream) {
    const float* cell    = (const float*)d_in[0];
    const float* hdr     = (const float*)d_in[1];
    const int*   ids     = (const int*)d_in[2];
    const int*   pos     = (const int*)d_in[3];
    const float* in_w    = (const float*)d_in[4];   // [2304,768]
    const float* in_b    = (const float*)d_in[5];   // [2304]
    const float* out_w   = (const float*)d_in[6];   // [768,768]
    const float* out_b   = (const float*)d_in[7];
    const float* row_emb = (const float*)d_in[8];
    const float* col_emb = (const float*)d_in[9];
    const float* fus_w   = (const float*)d_in[10];  // [768,1536]
    const float* fus_b   = (const float*)d_in[11];
    const float* ln_g    = (const float*)d_in[12];
    const float* ln_b    = (const float*)d_in[13];

    const int N  = in_sizes[0] / HIDDEN;     // 16384 cells
    const int NH = in_sizes[1] / HIDDEN;     // 512 headers

    // ---- workspace carve-up (all 256B aligned) ----
    char* ws = (char*)d_ws;
    size_t off = 0;
    auto carve = [&](size_t bytes) -> char* {
        char* p = ws + off;
        off += (bytes + 255) & ~(size_t)255;
        return p;
    };
    float*    cwp_f    = (float*)   carve((size_t)N * HIDDEN * 4);
    _Float16* cwp_h    = (_Float16*)carve((size_t)N * HIDDEN * 2);
    _Float16* q_h      = (_Float16*)carve((size_t)N * HIDDEN * 2);
    _Float16* ctx_h    = (_Float16*)carve((size_t)N * HIDDEN * 2);
    _Float16* attout_h = (_Float16*)carve((size_t)N * HIDDEN * 2);
    float*    khdr     = (float*)   carve((size_t)NH * HIDDEN * 4);
    float*    vhdr     = (float*)   carve((size_t)NH * HIDDEN * 4);
    _Float16* w_in_h   = (_Float16*)carve((size_t)3 * HIDDEN * HIDDEN * 2);
    _Float16* w_out_h  = (_Float16*)carve((size_t)HIDDEN * HIDDEN * 2);
    _Float16* w_fus_h  = (_Float16*)carve((size_t)HIDDEN * 2 * HIDDEN * 2);
    _Float16* hdr_h    = (_Float16*)carve((size_t)NH * HIDDEN * 2);
    (void)ws_size; (void)n_in; (void)out_size;

    // ---- 1) weight / header conversion to f16 ----
    {
        int n;
        n = 3 * HIDDEN * HIDDEN;
        cvt_f16_kernel<<<(n + 255) / 256, 256, 0, stream>>>(in_w, w_in_h, n);
        n = HIDDEN * HIDDEN;
        cvt_f16_kernel<<<(n + 255) / 256, 256, 0, stream>>>(out_w, w_out_h, n);
        n = HIDDEN * 2 * HIDDEN;
        cvt_f16_kernel<<<(n + 255) / 256, 256, 0, stream>>>(fus_w, w_fus_h, n);
        n = NH * HIDDEN;
        cvt_f16_kernel<<<(n + 255) / 256, 256, 0, stream>>>(hdr, hdr_h, n);
    }

    // ---- 2) positional add ----
    prep_kernel<<<N, 256, 0, stream>>>(cell, row_emb, col_emb, pos, cwp_f, cwp_h);

    // ---- 3) header K/V projections (512x768x768, f32 out) ----
    {
        dim3 g(NH / 64, HIDDEN / 128);
        gemm_kernel<false><<<g, 256, 0, stream>>>(
            hdr_h, w_in_h + (size_t)HIDDEN * HIDDEN, in_b + HIDDEN,
            khdr, HIDDEN, HIDDEN, HIDDEN, HIDDEN);
        gemm_kernel<false><<<g, 256, 0, stream>>>(
            hdr_h, w_in_h + (size_t)2 * HIDDEN * HIDDEN, in_b + 2 * HIDDEN,
            vhdr, HIDDEN, HIDDEN, HIDDEN, HIDDEN);
    }

    // ---- 4) Q projection (16384x768x768, f16 out) ----
    {
        dim3 g(N / 64, HIDDEN / 128);
        gemm_kernel<true><<<g, 256, 0, stream>>>(
            cwp_h, w_in_h, in_b, q_h, HIDDEN, HIDDEN, HIDDEN, HIDDEN);
    }

    // ---- 5) per-cell attention over gathered projected headers ----
    attn_kernel<<<N, 256, 0, stream>>>(q_h, khdr, vhdr, ids, ctx_h,
                                       (float*)d_out + (size_t)N * HIDDEN);

    // ---- 6) attention output projection (f16 out) ----
    {
        dim3 g(N / 64, HIDDEN / 128);
        gemm_kernel<true><<<g, 256, 0, stream>>>(
            ctx_h, w_out_h, out_b, attout_h, HIDDEN, HIDDEN, HIDDEN, HIDDEN);
    }

    // ---- 7) fused dual-GEMM + LayerNorm + ReLU + blend -> enriched ----
    fusion_kernel<<<N / 16, 512, 0, stream>>>(
        cwp_h, attout_h, w_fus_h, fus_b, ln_g, ln_b, cwp_f, ids, (float*)d_out);
}
